// GATNet_7653631721630
// MI455X (gfx1250) — compile-verified, matrix operand order checked by reading, not run on hardware
//
#include <hip/hip_runtime.h>
#include <math.h>

#define TPB 256
#define BM 64
#define BN 128
#define BK 32

// ---------------------------------------------------------------------------
// CDNA5 WMMA types (wave32): 16x16x32 bf16 -> f32 accumulate
// ---------------------------------------------------------------------------
typedef __attribute__((ext_vector_type(16))) __bf16 v16bf;
typedef __attribute__((ext_vector_type(8)))  float  v8f;

union ABFrag { v16bf v; unsigned w[8]; };

__device__ __forceinline__ unsigned short f2bf(float f) {
  unsigned u = __builtin_bit_cast(unsigned, f);
  u += 0x7FFFu + ((u >> 16) & 1u);   // round-to-nearest-even
  return (unsigned short)(u >> 16);
}
__device__ __forceinline__ unsigned packbf(float lo, float hi) {
  return (unsigned)f2bf(lo) | ((unsigned)f2bf(hi) << 16);
}
// monotonic (order-preserving) u32 encoding of float for atomicMax
__device__ __forceinline__ unsigned ordf(float f) {
  unsigned u = __builtin_bit_cast(unsigned, f);
  return (u & 0x80000000u) ? ~u : (u | 0x80000000u);
}
__device__ __forceinline__ float unordf(unsigned o) {
  unsigned u = (o & 0x80000000u) ? (o & 0x7FFFFFFFu) : ~o;
  return __builtin_bit_cast(float, u);
}
#define ORD_NEG_INF 0x007FFFFFu     // ordf(-inf)

// ---------------------------------------------------------------------------
// Tiled WMMA GEMM: C[M,N] = act(A[M,K] @ W[K,N] + bias)
// Block: 256 threads = 8 waves; block tile 64(M) x 128(N); K step 32.
// Wave (wm,wn) in 2x4 grid owns a 32x32 sub-tile = 2x2 WMMA tiles
// -> 4 v_wmma per wave per K step from 2 A + 2 B fragments.
// K loop is split into a branch-free interior main loop + one guarded tail.
// ACT: 0=none, 1=relu, 2=elu
// ---------------------------------------------------------------------------
template <int ACT>
__global__ void __launch_bounds__(TPB)
gemm_wmma(const float* __restrict__ A, const float* __restrict__ W,
          const float* __restrict__ bias, float* __restrict__ C,
          int M, int K, int N)
{
  __shared__ unsigned ldsA[BM][17];    // [m][kpair], padded vs bank conflicts
  __shared__ unsigned ldsB[BN][17];    // [n][kpair] (B transposed into LDS)

  const int tid  = threadIdx.x;
  const int m0   = blockIdx.y * BM;
  const int n0   = blockIdx.x * BN;
  const int lane = tid & 31;
  const int wv   = tid >> 5;
  const int wm   = wv >> 2;           // 0..1  -> 32-row slab
  const int wn   = wv & 3;            // 0..3  -> 32-col slab
  const int h16  = lane >> 4;         // lane half selects K sub-group
  const int l16  = lane & 15;

  v8f acc00 = {}, acc01 = {}, acc10 = {}, acc11 = {};

  const unsigned* rowA0 = &ldsA[wm * 32 + l16][0];
  const unsigned* rowA1 = &ldsA[wm * 32 + 16 + l16][0];
  const unsigned* rowB0 = &ldsB[wn * 32 + l16][0];
  const unsigned* rowB1 = &ldsB[wn * 32 + 16 + l16][0];

  // fragments per ISA 16-bit A 16x32 / B 32x16 lane layout, then 2x2 WMMAs
  auto compute = [&]() {
    ABFrag a0, a1, b0, b1;
#pragma unroll
    for (int v = 0; v < 8; ++v) {
      int kp = ((v >> 2) << 3) + (h16 << 2) + (v & 3);
      a0.w[v] = rowA0[kp];
      a1.w[v] = rowA1[kp];
      b0.w[v] = rowB0[kp];
      b1.w[v] = rowB1[kp];
    }
    acc00 = __builtin_amdgcn_wmma_f32_16x16x32_bf16(false, a0.v, false, b0.v,
                                                    (short)0, acc00, false, false);
    acc01 = __builtin_amdgcn_wmma_f32_16x16x32_bf16(false, a0.v, false, b1.v,
                                                    (short)0, acc01, false, false);
    acc10 = __builtin_amdgcn_wmma_f32_16x16x32_bf16(false, a1.v, false, b0.v,
                                                    (short)0, acc10, false, false);
    acc11 = __builtin_amdgcn_wmma_f32_16x16x32_bf16(false, a1.v, false, b1.v,
                                                    (short)0, acc11, false, false);
  };

  // fully guarded staging (M/N edge blocks and/or K tail)
  auto stage_guard = [&](int k0) {
    for (int p = tid; p < BM * 16; p += TPB) {
      int r = p >> 4, c = p & 15;
      int gm = m0 + r, gk = k0 + 2 * c;
      float f0 = 0.f, f1 = 0.f;
      if (gm < M) {
        if (gk     < K) f0 = A[(size_t)gm * K + gk];
        if (gk + 1 < K) f1 = A[(size_t)gm * K + gk + 1];
      }
      ldsA[r][c] = packbf(f0, f1);
    }
    for (int p = tid; p < BN * 16; p += TPB) {
      int n = p >> 4, c = p & 15;
      int gn = n0 + n, gk = k0 + 2 * c;
      float f0 = 0.f, f1 = 0.f;
      if (gn < N) {
        if (gk     < K) f0 = W[(size_t)gk * N + gn];
        if (gk + 1 < K) f1 = W[(size_t)(gk + 1) * N + gn];
      }
      ldsB[n][c] = packbf(f0, f1);
    }
  };

  const bool mFull = (m0 + BM <= M);
  const bool nFull = (n0 + BN <= N);
  const int  Ktail = K % BK;
  const int  Kmain = K - Ktail;

  if (mFull && nFull) {
    // branch-free interior main loop
    for (int k0 = 0; k0 < Kmain; k0 += BK) {
      // A: K is even for every layer -> float2 loads are 8B aligned
      for (int p = tid; p < BM * 16; p += TPB) {
        int r = p >> 4, c = p & 15;
        const float2 f = *(const float2*)(A + (size_t)(m0 + r) * K + k0 + 2 * c);
        ldsA[r][c] = packbf(f.x, f.y);
      }
      // B: N is a multiple of 4 for every layer -> float4 loads are 16B aligned
      for (int p = tid; p < 16 * (BN / 4); p += TPB) {
        int kp = p >> 5, nq = p & 31;                 // kpair, n-quad
        const float* base = W + (size_t)(k0 + 2 * kp) * N + n0 + 4 * nq;
        const float4 f0 = *(const float4*)base;
        const float4 f1 = *(const float4*)(base + N);
        ldsB[4 * nq + 0][kp] = packbf(f0.x, f1.x);
        ldsB[4 * nq + 1][kp] = packbf(f0.y, f1.y);
        ldsB[4 * nq + 2][kp] = packbf(f0.z, f1.z);
        ldsB[4 * nq + 3][kp] = packbf(f0.w, f1.w);
      }
      __syncthreads();
      compute();
      __syncthreads();
    }
  } else {
    for (int k0 = 0; k0 < Kmain; k0 += BK) {
      stage_guard(k0);
      __syncthreads();
      compute();
      __syncthreads();
    }
  }
  if (Ktail) {                          // single guarded K-tail step
    stage_guard(Kmain);
    __syncthreads();
    compute();
    __syncthreads();
  }

  // ---- store 2x2 tiles; C/D layout: n = lane&15, m = v + 8*(lane>>4) ----
  auto apply = [&](float x) {
    if (ACT == 1)      x = fmaxf(x, 0.f);
    else if (ACT == 2) x = (x > 0.f) ? x : (expf(x) - 1.f);
    return x;
  };
  const int tm = m0 + wm * 32 + h16 * 8;
  const int tn = n0 + wn * 32 + l16;
  if (mFull && nFull) {
    const float bv0 = bias ? bias[tn]      : 0.f;
    const float bv1 = bias ? bias[tn + 16] : 0.f;
#pragma unroll
    for (int v = 0; v < 8; ++v) {
      C[(size_t)(tm + v)      * N + tn]      = apply(acc00[v] + bv0);
      C[(size_t)(tm + v)      * N + tn + 16] = apply(acc01[v] + bv1);
      C[(size_t)(tm + 16 + v) * N + tn]      = apply(acc10[v] + bv0);
      C[(size_t)(tm + 16 + v) * N + tn + 16] = apply(acc11[v] + bv1);
    }
  } else {
    const float bv0 = (bias && tn      < N) ? bias[tn]      : 0.f;
    const float bv1 = (bias && tn + 16 < N) ? bias[tn + 16] : 0.f;
#pragma unroll
    for (int v = 0; v < 8; ++v) {
      int cm0 = tm + v, cm1 = tm + 16 + v;
      if (tn < N) {
        if (cm0 < M) C[(size_t)cm0 * N + tn] = apply(acc00[v] + bv0);
        if (cm1 < M) C[(size_t)cm1 * N + tn] = apply(acc10[v] + bv0);
      }
      if (tn + 16 < N) {
        if (cm0 < M) C[(size_t)cm0 * N + tn + 16] = apply(acc01[v] + bv1);
        if (cm1 < M) C[(size_t)cm1 * N + tn + 16] = apply(acc11[v] + bv1);
      }
    }
  }
}

// ---------------------------------------------------------------------------
// Elementwise / attention kernels
// ---------------------------------------------------------------------------
__global__ void fill_u32(unsigned* __restrict__ p, unsigned val, long long n) {
  long long i = (long long)blockIdx.x * blockDim.x + threadIdx.x;
  if (i < n) p[i] = val;
}

// alpha_s[n,h] = sum_c h[n,h,c]*a_src[h,c] ; alpha_d likewise
__global__ void attn_alpha(const float* __restrict__ hf,
                           const float* __restrict__ asrc,
                           const float* __restrict__ adst,
                           float* __restrict__ alps, float* __restrict__ alpd,
                           int N, int Hh, int Cc) {
  long long i = (long long)blockIdx.x * blockDim.x + threadIdx.x;
  if (i >= (long long)N * Hh) return;
  int hh = (int)(i % Hh);
  const float* hp = hf + (size_t)i * Cc;
  const float* sp = asrc + (size_t)hh * Cc;
  const float* dp = adst + (size_t)hh * Cc;
  float ss = 0.f, sd = 0.f;
  for (int c = 0; c < Cc; ++c) { float v = hp[c]; ss += v * sp[c]; sd += v * dp[c]; }
  alps[i] = ss; alpd[i] = sd;
}

// leaky_relu score + segment max over dst (ordered-int atomicMax)
__global__ void edge_max(const int* __restrict__ e0, const int* __restrict__ e1,
                         int E, int N, int Hh,
                         const float* __restrict__ alps, const float* __restrict__ alpd,
                         float* __restrict__ escore, unsigned* __restrict__ emaxb) {
  long long i = (long long)blockIdx.x * blockDim.x + threadIdx.x;
  long long tot = (long long)(E + N) * Hh;
  if (i >= tot) return;
  int e = (int)(i / Hh), hh = (int)(i % Hh);
  int s = (e < E) ? e0[e] : (e - E);
  int d = (e < E) ? e1[e] : (e - E);
  float x = alps[(size_t)s * Hh + hh] + alpd[(size_t)d * Hh + hh];
  x = (x > 0.f) ? x : 0.2f * x;
  escore[i] = x;
  atomicMax(&emaxb[(size_t)d * Hh + hh], ordf(x));
}

// ex = exp(score - max); denom[dst] += ex  (GLOBAL_ATOMIC_ADD_F32)
__global__ void edge_exp(const int* __restrict__ e1, int E, int N, int Hh,
                         const float* __restrict__ escore,
                         const unsigned* __restrict__ emaxb,
                         float* __restrict__ eex, float* __restrict__ denom) {
  long long i = (long long)blockIdx.x * blockDim.x + threadIdx.x;
  long long tot = (long long)(E + N) * Hh;
  if (i >= tot) return;
  int e = (int)(i / Hh), hh = (int)(i % Hh);
  int d = (e < E) ? e1[e] : (e - E);
  float ex = expf(escore[i] - unordf(emaxb[(size_t)d * Hh + hh]));
  eex[i] = ex;
  unsafeAtomicAdd(&denom[(size_t)d * Hh + hh], ex);
}

// aggr[dst,h,:] += h[src,h,:] * (ex / denom[dst,h])
__global__ void edge_msg(const int* __restrict__ e0, const int* __restrict__ e1,
                         int E, int N, int Hh, int Cc,
                         const float* __restrict__ hf, const float* __restrict__ eex,
                         const float* __restrict__ denom, float* __restrict__ aggr) {
  long long i = (long long)blockIdx.x * blockDim.x + threadIdx.x;
  long long tot = (long long)(E + N) * Hh;
  if (i >= tot) return;
  int e = (int)(i / Hh), hh = (int)(i % Hh);
  int s = (e < E) ? e0[e] : (e - E);
  int d = (e < E) ? e1[e] : (e - E);
  float alpha = eex[i] / (denom[(size_t)d * Hh + hh] + 1e-16f);
  const float* hp = hf + ((size_t)s * Hh + hh) * Cc;
  float* op = aggr + ((size_t)d * Hh + hh) * Cc;
  for (int c = 0; c < Cc; ++c) unsafeAtomicAdd(&op[c], hp[c] * alpha);
}

// x = elu(x + bias) in place
__global__ void bias_elu(float* __restrict__ x, const float* __restrict__ bias,
                         long long total, int F) {
  long long i = (long long)blockIdx.x * blockDim.x + threadIdx.x;
  if (i >= total) return;
  float v = x[i] + bias[(int)(i % F)];
  x[i] = (v > 0.f) ? v : (expf(v) - 1.f);
}

// global max pool over batch ids (ordered-int atomicMax)
__global__ void pool_max(const float* __restrict__ hf, const int* __restrict__ batch,
                         unsigned* __restrict__ gbits, int N, int Cc) {
  long long i = (long long)blockIdx.x * blockDim.x + threadIdx.x;
  if (i >= (long long)N * Cc) return;
  int n = (int)(i / Cc), c = (int)(i % Cc);
  atomicMax(&gbits[(size_t)batch[n] * Cc + c], ordf(hf[i]));
}

__global__ void decode_max(const unsigned* __restrict__ gbits, float* __restrict__ g,
                           long long total) {
  long long i = (long long)blockIdx.x * blockDim.x + threadIdx.x;
  if (i < total) g[i] = unordf(gbits[i]);
}

// row-wise L2 normalize (one block per row; in==out allowed)
__global__ void l2norm_rows(const float* __restrict__ in, float* __restrict__ out,
                            int cols) {
  __shared__ float red[TPB];
  const int row = blockIdx.x;
  const float* ip = in  + (size_t)row * cols;
  float*       op = out + (size_t)row * cols;
  float ss = 0.f;
  for (int c = threadIdx.x; c < cols; c += TPB) { float v = ip[c]; ss += v * v; }
  red[threadIdx.x] = ss; __syncthreads();
  for (int s = TPB / 2; s > 0; s >>= 1) {
    if (threadIdx.x < s) red[threadIdx.x] += red[threadIdx.x + s];
    __syncthreads();
  }
  float inv = 1.f / fmaxf(sqrtf(red[0]), 1e-12f);
  for (int c = threadIdx.x; c < cols; c += TPB) op[c] = ip[c] * inv;
}

// xc = [v1 | v2 | c]  (128 + 128 + 256 = 512)
__global__ void concat3(const float* __restrict__ a, const float* __restrict__ b,
                        const float* __restrict__ c, float* __restrict__ out, int Bn) {
  int i = blockIdx.x * blockDim.x + threadIdx.x;
  if (i >= Bn * 512) return;
  int row = i >> 9, col = i & 511;
  float v;
  if (col < 128)      v = a[row * 128 + col];
  else if (col < 256) v = b[row * 128 + (col - 128)];
  else                v = c[row * 256 + (col - 256)];
  out[i] = v;
}

// tiny final projection 128 -> 2
__global__ void final_out(const float* __restrict__ f3, const float* __restrict__ Wo,
                          const float* __restrict__ bo, float* __restrict__ out,
                          int Bn, int K, int O) {
  int i = blockIdx.x * blockDim.x + threadIdx.x;
  if (i >= Bn * O) return;
  int b = i / O, o = i % O;
  float s = bo[o];
  for (int k = 0; k < K; ++k) s += f3[(size_t)b * K + k] * Wo[(size_t)k * O + o];
  out[i] = s;
}

// ---------------------------------------------------------------------------
// Host orchestration
// ---------------------------------------------------------------------------
extern "C" void kernel_launch(void* const* d_in, const int* in_sizes, int n_in,
                              void* d_out, int out_size, void* d_ws, size_t ws_size,
                              hipStream_t stream) {
  (void)in_sizes; (void)n_in; (void)out_size; (void)ws_size;
  const int N = 30000, E = 100000, Bg = 1024;
  const int FXD = 78, FXT = 954, D = 128, H = 10;

  const float* x1     = (const float*)d_in[0];
  const int*   ei1p   = (const int*)d_in[1];
  const int*   batch1 = (const int*)d_in[2];
  const float* x2     = (const float*)d_in[3];
  const int*   ei2p   = (const int*)d_in[4];
  const int*   batch2 = (const int*)d_in[5];
  const float* cell   = (const float*)d_in[6];
  const float* W1     = (const float*)d_in[7];
  const float* a_src1 = (const float*)d_in[8];
  const float* a_dst1 = (const float*)d_in[9];
  const float* b1     = (const float*)d_in[10];
  const float* W2     = (const float*)d_in[11];
  const float* a_src2 = (const float*)d_in[12];
  const float* a_dst2 = (const float*)d_in[13];
  const float* b2     = (const float*)d_in[14];
  const float* Wg     = (const float*)d_in[15];
  const float* bg     = (const float*)d_in[16];
  const float* Wr1    = (const float*)d_in[17];
  const float* br1    = (const float*)d_in[18];
  const float* Wr2    = (const float*)d_in[19];
  const float* br2    = (const float*)d_in[20];
  const float* Wr3    = (const float*)d_in[21];
  const float* br3    = (const float*)d_in[22];
  const float* Wf1    = (const float*)d_in[23];
  const float* bf1    = (const float*)d_in[24];
  const float* Wf2    = (const float*)d_in[25];
  const float* bf2    = (const float*)d_in[26];
  const float* Wf3    = (const float*)d_in[27];
  const float* bf3    = (const float*)d_in[28];
  const float* Wo     = (const float*)d_in[29];
  const float* bo     = (const float*)d_in[30];

  // workspace carve-out
  char* wsp = (char*)d_ws;
  size_t off = 0;
  auto wsalloc = [&](size_t bytes) -> void* {
    void* p = wsp + off;
    off += (bytes + 255) & ~(size_t)255;
    return p;
  };
  float*    hfeat  = (float*)wsalloc((size_t)N * H * D * 4);   // transformed feats
  float*    aggr   = (float*)wsalloc((size_t)N * H * D * 4);   // aggregation buf
  float*    alps   = (float*)wsalloc((size_t)N * H * 4);
  float*    alpd   = (float*)wsalloc((size_t)N * H * 4);
  unsigned* emaxb  = (unsigned*)wsalloc((size_t)N * H * 4);
  float*    denom  = (float*)wsalloc((size_t)N * H * 4);
  float*    escore = (float*)wsalloc((size_t)(E + N) * H * 4);
  float*    eex    = (float*)wsalloc((size_t)(E + N) * H * 4);
  unsigned* gbits  = (unsigned*)wsalloc((size_t)Bg * D * 4);
  float*    gmax   = (float*)wsalloc((size_t)Bg * D * 4);
  float*    v1     = (float*)wsalloc((size_t)Bg * D * 4);
  float*    v2     = (float*)wsalloc((size_t)Bg * D * 4);
  float*    cn     = (float*)wsalloc((size_t)Bg * FXT * 4);
  float*    r1     = (float*)wsalloc((size_t)Bg * 2048 * 4);
  float*    r2     = (float*)wsalloc((size_t)Bg * 512 * 4);
  float*    r3     = (float*)wsalloc((size_t)Bg * 256 * 4);
  float*    xc     = (float*)wsalloc((size_t)Bg * 512 * 4);
  float*    f1     = (float*)wsalloc((size_t)Bg * 1024 * 4);
  float*    f2     = (float*)wsalloc((size_t)Bg * 512 * 4);
  float*    f3     = (float*)wsalloc((size_t)Bg * 128 * 4);

  auto nblk = [](long long n) { return (unsigned)((n + TPB - 1) / TPB); };
  auto fill = [&](void* p, unsigned val, long long n) {
    fill_u32<<<nblk(n), TPB, 0, stream>>>((unsigned*)p, val, n);
  };
  auto gemm = [&](int act, const float* A, const float* Wt, const float* bs,
                  float* Cp, int M, int K, int Nn) {
    dim3 g((unsigned)((Nn + BN - 1) / BN), (unsigned)((M + BM - 1) / BM));
    if (act == 0)      gemm_wmma<0><<<g, TPB, 0, stream>>>(A, Wt, bs, Cp, M, K, Nn);
    else if (act == 1) gemm_wmma<1><<<g, TPB, 0, stream>>>(A, Wt, bs, Cp, M, K, Nn);
    else               gemm_wmma<2><<<g, TPB, 0, stream>>>(A, Wt, bs, Cp, M, K, Nn);
  };

  auto branch = [&](const float* x, const int* ei, const int* batch, float* vout) {
    const int* e0 = ei;
    const int* e1 = ei + E;
    const long long teH  = (long long)(E + N) * H;
    const long long te1  = (long long)(E + N);

    // ---- GATConv1: 78 -> 10x128 (concat) ----
    gemm(0, x, W1, nullptr, hfeat, N, FXD, H * D);
    fill(aggr, 0u, (long long)N * H * D);
    fill(emaxb, ORD_NEG_INF, (long long)N * H);
    fill(denom, 0u, (long long)N * H);
    attn_alpha<<<nblk((long long)N * H), TPB, 0, stream>>>(hfeat, a_src1, a_dst1,
                                                           alps, alpd, N, H, D);
    edge_max<<<nblk(teH), TPB, 0, stream>>>(e0, e1, E, N, H, alps, alpd, escore, emaxb);
    edge_exp<<<nblk(teH), TPB, 0, stream>>>(e1, E, N, H, escore, emaxb, eex, denom);
    edge_msg<<<nblk(teH), TPB, 0, stream>>>(e0, e1, E, N, H, D, hfeat, eex, denom, aggr);
    bias_elu<<<nblk((long long)N * H * D), TPB, 0, stream>>>(aggr, b1,
                                                             (long long)N * H * D, H * D);

    // ---- GATConv2: 1280 -> 128 (1 head) ----
    gemm(0, aggr, W2, nullptr, hfeat, N, H * D, D);   // hfeat[:N*128] = h1 @ W2
    fill(aggr, 0u, (long long)N * D);                 // reuse aggr for conv2 sums
    fill(emaxb, ORD_NEG_INF, (long long)N);
    fill(denom, 0u, (long long)N);
    attn_alpha<<<nblk((long long)N), TPB, 0, stream>>>(hfeat, a_src2, a_dst2,
                                                       alps, alpd, N, 1, D);
    edge_max<<<nblk(te1), TPB, 0, stream>>>(e0, e1, E, N, 1, alps, alpd, escore, emaxb);
    edge_exp<<<nblk(te1), TPB, 0, stream>>>(e1, E, N, 1, escore, emaxb, eex, denom);
    edge_msg<<<nblk(te1), TPB, 0, stream>>>(e0, e1, E, N, 1, D, hfeat, eex, denom, aggr);
    bias_elu<<<nblk((long long)N * D), TPB, 0, stream>>>(aggr, b2, (long long)N * D, D);

    // ---- global max pool + relu(g @ Wg + bg) ----
    fill(gbits, ORD_NEG_INF, (long long)Bg * D);
    pool_max<<<nblk((long long)N * D), TPB, 0, stream>>>(aggr, batch, gbits, N, D);
    decode_max<<<nblk((long long)Bg * D), TPB, 0, stream>>>(gbits, gmax, (long long)Bg * D);
    gemm(1, gmax, Wg, bg, vout, Bg, D, D);
  };

  branch(x1, ei1p, batch1, v1);
  branch(x2, ei2p, batch2, v2);

  // ---- cell MLP ----
  l2norm_rows<<<Bg, TPB, 0, stream>>>(cell, cn, FXT);
  gemm(1, cn, Wr1, br1, r1, Bg, FXT, 2048);
  gemm(1, r1, Wr2, br2, r2, Bg, 2048, 512);
  gemm(1, r2, Wr3, br3, r3, Bg, 512, 256);

  // ---- fusion head ----
  concat3<<<nblk((long long)Bg * 512), TPB, 0, stream>>>(v1, v2, r3, xc, Bg);
  l2norm_rows<<<Bg, TPB, 0, stream>>>(xc, xc, 512);
  gemm(1, xc, Wf1, bf1, f1, Bg, 512, 1024);
  gemm(1, f1, Wf2, bf2, f2, Bg, 1024, 512);
  gemm(1, f2, Wf3, bf3, f3, Bg, 512, 128);
  final_out<<<nblk((long long)Bg * 2), TPB, 0, stream>>>(f3, Wo, bo, (float*)d_out,
                                                         Bg, 128, 2);
}
